// MultiHeadedSelfAttention_26010321944947
// MI455X (gfx1250) — matrix operand order repaired and checked by reading, not compiled, not run on hardware
//
#include <hip/hip_runtime.h>

// ---------------------------------------------------------------------------
// Multi-head self-attention for MI455X (gfx1250, wave32, WMMA).
//   B=4, N=2048, D=1024, KS=OS=1024, H=16, head dim W=64.
// Kernel 1: fused q/k/v/c projections (bf16 WMMA GEMM, fp32 accumulate).
// Kernel 2: flash-attention (QK^T + online softmax + PV), bf16 WMMA,
//           K/V staged global->LDS via async-load path when available.
// ---------------------------------------------------------------------------

typedef __attribute__((ext_vector_type(16))) __bf16 v16bf;
typedef __attribute__((ext_vector_type(8)))  float  v8f;

union Frag {
  v16bf v;
  uint4 q[2];
};

constexpr int Bn  = 4;
constexpr int Nn  = 2048;
constexpr int Dn  = 1024;
constexpr int KSn = 1024;
constexpr int Hn  = 16;

#define DEV __device__ __forceinline__

#if defined(__gfx1250__) &&                                              \
    __has_builtin(__builtin_amdgcn_global_load_async_to_lds_b128) &&     \
    __has_builtin(__builtin_amdgcn_s_wait_asynccnt)
#define USE_ASYNC_LDS 1
#else
#define USE_ASYNC_LDS 0
#endif

#if USE_ASYNC_LDS
typedef int v4i_t __attribute__((ext_vector_type(4)));
typedef __attribute__((address_space(1))) v4i_t* gvp_t;  // global int4*
typedef __attribute__((address_space(3))) v4i_t* lvp_t;  // LDS int4*
DEV void async_copy16(const void* g, void* l) {
  // (global src, lds dst, imm offset, imm cpol) -- 16B per lane, ASYNCcnt.
  __builtin_amdgcn_global_load_async_to_lds_b128((gvp_t)(void*)g, (lvp_t)l, 0, 0);
}
#endif

DEV unsigned short f2bf_bits(float f) {
  unsigned u = __builtin_bit_cast(unsigned, f);
  unsigned r = u + 0x7FFFu + ((u >> 16) & 1u);   // round-to-nearest-even
  return (unsigned short)(r >> 16);
}
DEV __bf16 f2bf(float f) {
  unsigned short h = f2bf_bits(f);
  return __builtin_bit_cast(__bf16, h);
}
DEV unsigned pack2(float a, float b) {
  return (unsigned)f2bf_bits(a) | ((unsigned)f2bf_bits(b) << 16);
}
DEV v8f wmma_bf16(v16bf a, v16bf b, v8f c) {
  // (neg_a, A, neg_b, B, c_mod, C, reuse_a, reuse_b)
  return __builtin_amdgcn_wmma_f32_16x16x32_bf16(false, a, false, b,
                                                 (short)0, c, false, false);
}

// ---------------------------------------------------------------------------
// Kernel 1: projections.  grid = (M/64, KS/128, 4), block = 256 (8 waves).
// Wave w computes a 16x64 strip: wm = w&3 row strip, wn = w>>2 column half.
// z = 0:q  1:k  2:v  3:c
// ---------------------------------------------------------------------------
__global__ __launch_bounds__(256) void mhsa_proj_kernel(
    const float* __restrict__ x,
    const float* __restrict__ Wq, const float* __restrict__ bq,
    const float* __restrict__ Wk, const float* __restrict__ bk,
    const float* __restrict__ Wv, const float* __restrict__ bv,
    const float* __restrict__ Wc, const float* __restrict__ bc,
    __bf16* __restrict__ qws, __bf16* __restrict__ kws,
    __bf16* __restrict__ vws, float* __restrict__ outc)
{
  __shared__ __align__(16) __bf16 ldsA[64 * 32];    // x tile  [row][k]
  __shared__ __align__(16) __bf16 ldsBT[128 * 32];  // W tile  [col][k] (transposed)

  const int tid   = threadIdx.x;
  const int wave  = tid >> 5;
  const int lane  = tid & 31;
  const int hi    = (lane >= 16) ? 1 : 0;
  const int col16 = lane & 15;
  const int wm = wave & 3;
  const int wn = wave >> 2;
  const int m0 = blockIdx.x * 64;
  const int n0 = blockIdx.y * 128;
  const int z  = blockIdx.z;

  const float* Wt = (z == 0) ? Wq : (z == 1) ? Wk : (z == 2) ? Wv : Wc;
  const float* bt = (z == 0) ? bq : (z == 1) ? bk : (z == 2) ? bv : bc;

  // staging coordinates (computed once)
  const int ar  = (tid * 8) >> 5;          // A row        0..63
  const int akc = (tid * 8) & 31;          // A k offset   {0,8,16,24}
  const int bk2 = (tid & 15) * 2;          // B k pair     0..30 (even)
  const int bc0 = (tid >> 4) * 8;          // B col group  0..120

  v8f acc[4] = {};

  for (int kt = 0; kt < Dn / 32; ++kt) {
    __syncthreads();
    // --- stage A (x) tile: 64x32 f32 -> bf16, 8 elems/thread ---------------
    {
      const float4* g = (const float4*)(x + (size_t)(m0 + ar) * Dn + kt * 32 + akc);
      float4 f0 = g[0], f1 = g[1];
      uint4 st;
      st.x = pack2(f0.x, f0.y); st.y = pack2(f0.z, f0.w);
      st.z = pack2(f1.x, f1.y); st.w = pack2(f1.z, f1.w);
      *(uint4*)(ldsA + ar * 32 + akc) = st;
      if (kt + 1 < Dn / 32)
        __builtin_prefetch(x + (size_t)(m0 + ar) * Dn + (kt + 1) * 32 + akc, 0, 1);
    }
    // --- stage B (W) tile: 32x128 f32 -> bf16, transposed [col][k], packed --
    {
      const float* g0 = Wt + (size_t)(kt * 32 + bk2) * KSn + n0 + bc0;
      const float* g1 = g0 + KSn;
      float4 a0 = *(const float4*)(g0);
      float4 a1 = *(const float4*)(g0 + 4);
      float4 b0 = *(const float4*)(g1);
      float4 b1 = *(const float4*)(g1 + 4);
      unsigned* lb = (unsigned*)ldsBT;
      // element (c, kc) lives at c*32 + kc ; pack (kc, kc+1) into one b32
      lb[((bc0 + 0) * 32 + bk2) >> 1] = pack2(a0.x, b0.x);
      lb[((bc0 + 1) * 32 + bk2) >> 1] = pack2(a0.y, b0.y);
      lb[((bc0 + 2) * 32 + bk2) >> 1] = pack2(a0.z, b0.z);
      lb[((bc0 + 3) * 32 + bk2) >> 1] = pack2(a0.w, b0.w);
      lb[((bc0 + 4) * 32 + bk2) >> 1] = pack2(a1.x, b1.x);
      lb[((bc0 + 5) * 32 + bk2) >> 1] = pack2(a1.y, b1.y);
      lb[((bc0 + 6) * 32 + bk2) >> 1] = pack2(a1.z, b1.z);
      lb[((bc0 + 7) * 32 + bk2) >> 1] = pack2(a1.w, b1.w);
      if (kt + 1 < Dn / 32)
        __builtin_prefetch(g0 + (size_t)32 * KSn, 0, 1);
    }
    __syncthreads();

    // --- batch all fragment loads, then the 4 WMMAs ------------------------
    Frag fa, fb[4];
    {
      int row = wm * 16 + col16;
      int b8  = hi ? 8 : 0;
      fa.q[0] = *(const uint4*)(ldsA + row * 32 + b8);
      fa.q[1] = *(const uint4*)(ldsA + row * 32 + 16 + b8);
    }
    const int kb = hi ? 8 : 0;
#pragma unroll
    for (int sub = 0; sub < 4; ++sub) {
      int c = wn * 64 + sub * 16 + col16;
      const uint4* p = (const uint4*)(ldsBT + c * 32 + kb);
      fb[sub].q[0] = p[0];
      fb[sub].q[1] = p[2];
    }
#pragma unroll
    for (int sub = 0; sub < 4; ++sub)
      acc[sub] = wmma_bf16(fa.v, fb[sub].v, acc[sub]);
  }

  // --- epilogue: bias + route to q/k/v/c layouts ---------------------------
  constexpr float QSC = 0.125f * 1.4426950408889634f;  // (1/sqrt(64))*log2(e)
#pragma unroll
  for (int sub = 0; sub < 4; ++sub) {
    int ncol = n0 + wn * 64 + sub * 16 + col16;
    float bval = bt[ncol];
    int hh = ncol >> 6;
    int ww = ncol & 63;
#pragma unroll
    for (int r = 0; r < 8; ++r) {
      int m = m0 + wm * 16 + r + hi * 8;
      float val = acc[sub][r] + bval;
      int bb = m >> 11;          // / N
      int nr = m & (Nn - 1);     // % N
      if (z == 0) {
        qws[(((size_t)bb * Hn + hh) * Nn + nr) * 64 + ww] = f2bf(val * QSC);
      } else if (z == 1) {
        kws[(((size_t)bb * Hn + hh) * Nn + nr) * 64 + ww] = f2bf(val);
      } else if (z == 2) {
        vws[(((size_t)bb * Hn + hh) * 64 + ww) * Nn + nr] = f2bf(val);  // transposed
      } else {
        outc[(size_t)m * KSn + ncol] = val;  // residual c, fp32
      }
    }
  }
}

// ---------------------------------------------------------------------------
// Kernel 2: flash attention.  grid = (N/64, B*H), block = 128 (4 waves).
// Each wave owns 16 query rows; iterates over 32 key blocks of 64.
// ---------------------------------------------------------------------------
__global__ __launch_bounds__(128) void mhsa_attn_kernel(
    const __bf16* __restrict__ qws, const __bf16* __restrict__ kws,
    const __bf16* __restrict__ vws, float* __restrict__ outc)
{
  __shared__ __align__(16) __bf16 ldsK[64 * 64];      // [key][w]
  __shared__ __align__(16) __bf16 ldsV[64 * 64];      // [w][key]
  __shared__ __align__(16) __bf16 ldsP[4 * 16 * 64];  // per-wave P slots

  const int tid   = threadIdx.x;
  const int wave  = tid >> 5;
  const int lane  = tid & 31;
  const int hi    = (lane >= 16) ? 1 : 0;
  const int col16 = lane & 15;
  const int bh    = blockIdx.y;
  const int qrow0 = blockIdx.x * 64 + wave * 16;

  const __bf16* qp = qws + (size_t)bh * Nn * 64;
  const __bf16* kp = kws + (size_t)bh * Nn * 64;
  const __bf16* vp = vws + (size_t)bh * 64 * Nn;

  // Q fragments (persistent): A-layout, K split {b..b+7, 16+b..16+b+7}
  Frag qf[2];
  {
    const __bf16* qr = qp + (size_t)(qrow0 + col16) * 64;
    int b8 = hi ? 8 : 0;
#pragma unroll
    for (int kk = 0; kk < 2; ++kk) {
      qf[kk].q[0] = *(const uint4*)(qr + kk * 32 + b8);
      qf[kk].q[1] = *(const uint4*)(qr + kk * 32 + b8 + 16);
    }
  }

  v8f acc[4] = {};
  float mr[8], lr[8];
#pragma unroll
  for (int r = 0; r < 8; ++r) { mr[r] = -3.0e38f; lr[r] = 0.0f; }

  __bf16* pslot = ldsP + wave * 16 * 64;
  const int srow  = tid >> 1;   // staging row   0..63
  const int shalf = tid & 1;    // staging half  0..1

  for (int kb = 0; kb < Nn / 64; ++kb) {
    const int key0 = kb * 64;
    __syncthreads();
    // --- stage K/V block (bf16, V pre-transposed): 2x16B per thread --------
    {
      const __bf16* gk = kp + (size_t)(key0 + srow) * 64 + shalf * 32;
      const __bf16* gv = vp + (size_t)srow * Nn + key0 + shalf * 32;
      __bf16* lk = ldsK + srow * 64 + shalf * 32;
      __bf16* lv = ldsV + srow * 64 + shalf * 32;
#if USE_ASYNC_LDS
      async_copy16(gk,      lk);
      async_copy16(gk + 16, lk + 16);
      async_copy16(gv,      lv);
      async_copy16(gv + 16, lv + 16);
#else
      ((uint4*)lk)[0] = ((const uint4*)gk)[0];
      ((uint4*)lk)[1] = ((const uint4*)gk)[1];
      ((uint4*)lv)[0] = ((const uint4*)gv)[0];
      ((uint4*)lv)[1] = ((const uint4*)gv)[1];
#endif
      if (kb + 1 < Nn / 64) {
        __builtin_prefetch(gk + 64, 0, 1);      // next key row, same half
        __builtin_prefetch(gv + 64, 0, 1);      // next V block, same w row
      }
    }
#if USE_ASYNC_LDS
    __builtin_amdgcn_s_wait_asynccnt(0);
#endif
    __syncthreads();

    // --- S = Q @ K^T : per k-step load 4 B-fragments, then 4 WMMAs ---------
    v8f s[4] = {};
#pragma unroll
    for (int kk = 0; kk < 2; ++kk) {
      Frag fb[4];
      const int wb = kk * 32 + (hi ? 8 : 0);
#pragma unroll
      for (int nt = 0; nt < 4; ++nt) {
        const uint4* p = (const uint4*)(ldsK + (nt * 16 + col16) * 64 + wb);
        fb[nt].q[0] = p[0];
        fb[nt].q[1] = p[2];
      }
#pragma unroll
      for (int nt = 0; nt < 4; ++nt)
        s[nt] = wmma_bf16(qf[kk].v, fb[nt].v, s[nt]);
    }

    // --- online softmax (base-2; scale folded into q) ----------------------
    float rmax[8], rsum[8], sc[8];
#pragma unroll
    for (int r = 0; r < 8; ++r)
      rmax[r] = fmaxf(fmaxf(s[0][r], s[1][r]), fmaxf(s[2][r], s[3][r]));
#pragma unroll
    for (int r = 0; r < 8; ++r) {
#pragma unroll
      for (int msk = 8; msk >= 1; msk >>= 1)
        rmax[r] = fmaxf(rmax[r], __shfl_xor(rmax[r], msk, 32));
      float mnew = fmaxf(mr[r], rmax[r]);
      sc[r] = exp2f(mr[r] - mnew);
      mr[r] = mnew;
      rsum[r] = 0.0f;
    }
#pragma unroll
    for (int nt = 0; nt < 4; ++nt) {
#pragma unroll
      for (int r = 0; r < 8; ++r) {
        float pv = exp2f(s[nt][r] - mr[r]);
        rsum[r] += pv;
        pslot[(r + hi * 8) * 64 + nt * 16 + col16] = f2bf(pv);  // C-layout -> LDS
      }
    }
#pragma unroll
    for (int r = 0; r < 8; ++r) {
#pragma unroll
      for (int msk = 8; msk >= 1; msk >>= 1)
        rsum[r] += __shfl_xor(rsum[r], msk, 32);
      lr[r] = lr[r] * sc[r] + rsum[r];
    }
#pragma unroll
    for (int nw = 0; nw < 4; ++nw)
#pragma unroll
      for (int r = 0; r < 8; ++r)
        acc[nw][r] *= sc[r];

    // --- reload P as A-fragments (wave-private LDS slot) -------------------
    Frag pf[2];
    {
      int b8 = hi ? 8 : 0;
#pragma unroll
      for (int kk = 0; kk < 2; ++kk) {
        int kbase = kk * 32 + b8;
        pf[kk].q[0] = *(const uint4*)(pslot + col16 * 64 + kbase);
        pf[kk].q[1] = *(const uint4*)(pslot + col16 * 64 + kbase + 16);
      }
    }
    // --- O += P @ V : per k-step load 4 B-fragments, then 4 WMMAs ----------
#pragma unroll
    for (int kk = 0; kk < 2; ++kk) {
      Frag fb[4];
      const int keyb = kk * 32 + (hi ? 8 : 0);
#pragma unroll
      for (int nw = 0; nw < 4; ++nw) {
        const uint4* p = (const uint4*)(ldsV + (nw * 16 + col16) * 64 + keyb);
        fb[nw].q[0] = p[0];
        fb[nw].q[1] = p[2];
      }
#pragma unroll
      for (int nw = 0; nw < 4; ++nw)
        acc[nw] = wmma_bf16(pf[kk].v, fb[nw].v, acc[nw]);
    }
  }

  // --- epilogue: normalize and add residual c (already in d_out) -----------
  const int b = bh / Hn;
  const int h = bh % Hn;
  float inv[8];
#pragma unroll
  for (int r = 0; r < 8; ++r) inv[r] = 1.0f / lr[r];
#pragma unroll
  for (int nw = 0; nw < 4; ++nw) {
    int ocol = h * 64 + nw * 16 + col16;
#pragma unroll
    for (int r = 0; r < 8; ++r) {
      int row = qrow0 + r + hi * 8;
      size_t idx = ((size_t)b * Nn + row) * KSn + ocol;
      outc[idx] = outc[idx] + acc[nw][r] * inv[r];
    }
  }
}

// ---------------------------------------------------------------------------
extern "C" void kernel_launch(void* const* d_in, const int* in_sizes, int n_in,
                              void* d_out, int out_size, void* d_ws, size_t ws_size,
                              hipStream_t stream) {
  (void)in_sizes; (void)n_in; (void)out_size; (void)ws_size;
  const float* x  = (const float*)d_in[0];
  const float* Wq = (const float*)d_in[1];
  const float* bq = (const float*)d_in[2];
  const float* Wk = (const float*)d_in[3];
  const float* bk = (const float*)d_in[4];
  const float* Wv = (const float*)d_in[5];
  const float* bv = (const float*)d_in[6];
  const float* Wc = (const float*)d_in[7];
  const float* bc = (const float*)d_in[8];
  float* outc = (float*)d_out;

  const size_t per = (size_t)Bn * Hn * Nn * 64;  // 8M bf16 elems per tensor
  __bf16* qws = (__bf16*)d_ws;
  __bf16* kws = qws + per;
  __bf16* vws = kws + per;

  dim3 gp((Bn * Nn) / 64, KSn / 128, 4);
  mhsa_proj_kernel<<<gp, 256, 0, stream>>>(x, Wq, bq, Wk, bk, Wv, bv, Wc, bc,
                                           qws, kws, vws, outc);
  dim3 ga(Nn / 64, Bn * Hn);
  mhsa_attn_kernel<<<ga, 128, 0, stream>>>(qws, kws, vws, outc);
}
